// Re_Attention_88751204205239
// MI455X (gfx1250) — matrix operand order repaired and checked by reading, not compiled, not run on hardware
//
#include <hip/hip_runtime.h>
#include <hip/hip_bf16.h>

typedef __attribute__((ext_vector_type(16))) _Float16 v16h;
typedef __attribute__((ext_vector_type(8)))  _Float16 v8h;
typedef __attribute__((ext_vector_type(8)))  float    v8f;

union V16U { v16h v; v8h h[2]; };

#define DIMC 512
#define HEADS 8
#define DHEAD 64
#define SEQ 1024
#define BATCH 8
#define MTOT (BATCH * SEQ)          // 8192
#define QKVN (3 * DIMC)             // 1536

__device__ __forceinline__ v8f wmma_f16(v16h a, v16h b, v8f c) {
    return __builtin_amdgcn_wmma_f32_16x16x32_f16(false, a, false, b, (short)0, c, false, false);
}

// A fragment (16x32 f16) from row-major [.., ld] at rows m0..m0+15, K = k0..k0+31.
// Layout: lanes 0-15 -> M=lane, K 0..15;
//         lanes 16-31 -> M=lane-16, K 16..31.
__device__ __forceinline__ v16h load_frag_a(const _Float16* p, int m0, int k0, int ld) {
    int lane = threadIdx.x & 31;
    const _Float16* s = p + (size_t)(m0 + (lane & 15)) * ld + k0 + ((lane >> 4) << 4);
    V16U u;
    u.h[0] = *(const v8h*)(s);
    u.h[1] = *(const v8h*)(s + 8);
    return u.v;
}

// B fragment (32x16 f16) where memory holds B^T row-major [N, ldk]
// (element (k,n) at p[n*ldk + k]).  Layout: lanes 0-15 -> N=lane, halves = K 0..15;
// lanes 16-31 -> N=lane-16, halves = K 16..31.  Fully contiguous 32B per lane.
__device__ __forceinline__ v16h load_frag_bT(const _Float16* p, int k0, int n0, int ldk) {
    int lane = threadIdx.x & 31;
    const _Float16* s = p + (size_t)(n0 + (lane & 15)) * ldk + k0 + ((lane >> 4) << 4);
    return *(const v16h*)s;
}

// ---------------- conversion / transpose prep kernels ----------------
__global__ void cvt_f16(const float* __restrict__ src, _Float16* __restrict__ dst, int n) {
    int i = blockIdx.x * blockDim.x + threadIdx.x;
    if (i < n) dst[i] = (_Float16)src[i];
}

__global__ void tcvt_f16(const float* __restrict__ src, _Float16* __restrict__ dst,
                         int rows, int cols) {   // src[rows][cols] f32 -> dst[cols][rows] f16
    int i = blockIdx.x * blockDim.x + threadIdx.x;
    if (i < rows * cols) {
        int r = i / cols, c = i - r * cols;
        dst[(size_t)c * rows + r] = (_Float16)src[i];
    }
}

// ---------------- QKV GEMM: xh[8192,512] @ wqkvT -> Q/K/V [B,H,N,D] f16 --------
// Each wave computes a 16x64 strip (A fragment reused across 4 B tiles).
__global__ void qkv_gemm(const _Float16* __restrict__ xh, const _Float16* __restrict__ wT,
                         _Float16* __restrict__ Qb, _Float16* __restrict__ Kb,
                         _Float16* __restrict__ Vb) {
    int wave = threadIdx.x >> 5;
    int lane = threadIdx.x & 31;
    int m0 = blockIdx.x * 16;
    int n0 = blockIdx.y * 512 + wave * 64;
    v8f acc[4] = {{}, {}, {}, {}};
#pragma unroll 2
    for (int k0 = 0; k0 < DIMC; k0 += 32) {
        v16h a = load_frag_a(xh, m0, k0, DIMC);
#pragma unroll
        for (int dt = 0; dt < 4; ++dt) {
            v16h b = load_frag_bT(wT, k0, n0 + dt * 16, DIMC);
            acc[dt] = wmma_f16(a, b, acc[dt]);
        }
    }
    int mbase = m0 + ((lane >> 4) << 3);
#pragma unroll
    for (int dt = 0; dt < 4; ++dt) {
        int col = n0 + dt * 16 + (lane & 15);
        int which = col >> 9;              // 0=q 1=k 2=v
        int h = (col >> 6) & 7;
        int d = col & 63;
        float scl = (which == 0) ? 0.125f : 1.0f;   // DIM_HEAD^-0.5 baked into Q
        _Float16* dst = (which == 0) ? Qb : ((which == 1) ? Kb : Vb);
#pragma unroll
        for (int r = 0; r < 8; ++r) {
            int m = mbase + r;
            int bb = m >> 10, n = m & 1023;
            dst[(((size_t)(bb * HEADS + h)) * SEQ + n) * DHEAD + d] =
                (_Float16)(acc[dt][r] * scl);
        }
    }
}

// ---------------- fused re-attention core ----------------
// block = 256 threads = 8 waves (wave == head), grid = (N/16, B)
__global__ void __launch_bounds__(256)
reattn_core(const _Float16* __restrict__ Qb, const _Float16* __restrict__ Kb,
            const _Float16* __restrict__ Vb, const float* __restrict__ reattn,
            const float* __restrict__ ln_g, const float* __restrict__ ln_b,
            _Float16* __restrict__ attn_out) {
    __shared__ float lds_p[HEADS][32][16];                     // softmaxed p, [h][j][i]
    __shared__ __align__(32) _Float16 lds_a[HEADS][16][32];    // remixed+LN,  [g][i][j]
    __shared__ __align__(16) _Float16 lds_v[HEADS][32][64];    // async-staged V tiles
    __shared__ float lds_W[HEADS][HEADS];
    __shared__ float lds_gam[HEADS], lds_bet[HEADS];

    int tid = threadIdx.x;
    int h = tid >> 5;                // wave id == head
    int lane = tid & 31;
    int i0 = blockIdx.x * 16;
    int b = blockIdx.y;

    if (tid < 64) lds_W[tid >> 3][tid & 7] = reattn[tid];
    if (tid < 8) { lds_gam[tid] = ln_g[tid]; lds_bet[tid] = ln_b[tid]; }
    __syncthreads();

    const _Float16* Qh = Qb + ((size_t)(b * HEADS + h)) * SEQ * DHEAD;
    const _Float16* Kh = Kb + ((size_t)(b * HEADS + h)) * SEQ * DHEAD;
    const _Float16* Vh = Vb + ((size_t)(b * HEADS + h)) * SEQ * DHEAD;

    // Q^T B-fragments, resident for the whole kernel (cols = i, K = d)
    v16h bq0 = load_frag_bT(Qh, 0, i0, DHEAD);
    v16h bq1 = load_frag_bT(Qh, 32, i0, DHEAD);

    // ---- pass 1: online softmax row stats (per i, reduced over all j) ----
    float mrow = -3.0e38f, lrow = 0.0f;
    for (int j0 = 0; j0 < SEQ; j0 += 16) {
        v16h a0 = load_frag_a(Kh, j0, 0, DHEAD);
        v16h a1 = load_frag_a(Kh, j0, 32, DHEAD);
        v8f z = {};
        v8f s = wmma_f16(a1, bq1, wmma_f16(a0, bq0, z));
        float tm = s[0];
#pragma unroll
        for (int r = 1; r < 8; ++r) tm = fmaxf(tm, s[r]);
        float nm = fmaxf(mrow, tm);
        float racc = 0.0f;
#pragma unroll
        for (int r = 0; r < 8; ++r) racc += __expf(s[r] - nm);
        lrow = lrow * __expf(mrow - nm) + racc;
        mrow = nm;
    }
    // combine the two lanes that own column i (lane and lane^16)
    float mo = __shfl_xor(mrow, 16);
    float lo = __shfl_xor(lrow, 16);
    float mf = fmaxf(mrow, mo);
    float inv_l = 1.0f / (lrow * __expf(mrow - mf) + lo * __expf(mo - mf));

    // ---- pass 2: p -> remix -> head-LN -> p@V, 32 j's at a time ----
    v8f accd[4] = {{}, {}, {}, {}};
    for (int j0 = 0; j0 < SEQ; j0 += 32) {
        // Kick off async DMA of this wave's V tile (32 rows x 64d, 4KB) into LDS.
        // One lane per row; 8 x b128 per lane row. Tracked on ASYNCcnt; overlaps
        // with the S^T WMMAs + softmax + remix below. Only wave h touches lds_v[h].
        {
            unsigned ldsa = (unsigned)(size_t)&lds_v[h][lane][0];
            const _Float16* g = Vh + (size_t)(j0 + lane) * DHEAD;
            asm volatile(
                "global_load_async_to_lds_b128 %0, %1, off\n\t"
                "global_load_async_to_lds_b128 %0, %1, off offset:16\n\t"
                "global_load_async_to_lds_b128 %0, %1, off offset:32\n\t"
                "global_load_async_to_lds_b128 %0, %1, off offset:48\n\t"
                "global_load_async_to_lds_b128 %0, %1, off offset:64\n\t"
                "global_load_async_to_lds_b128 %0, %1, off offset:80\n\t"
                "global_load_async_to_lds_b128 %0, %1, off offset:96\n\t"
                "global_load_async_to_lds_b128 %0, %1, off offset:112"
                :: "v"(ldsa), "v"(g) : "memory");
        }

#pragma unroll
        for (int jt = 0; jt < 2; ++jt) {
            int jj = j0 + jt * 16;
            v16h a0 = load_frag_a(Kh, jj, 0, DHEAD);
            v16h a1 = load_frag_a(Kh, jj, 32, DHEAD);
            v8f z = {};
            v8f s = wmma_f16(a1, bq1, wmma_f16(a0, bq0, z));
            int ic = lane & 15;
            int jbase = jt * 16 + ((lane >> 4) << 3);
#pragma unroll
            for (int r = 0; r < 8; ++r)
                lds_p[h][jbase + r][ic] = __expf(s[r] - mf) * inv_l;
        }
        __syncthreads();

        // remix across heads + LayerNorm over head axis; 512 (j,i) pairs / 256 threads
#pragma unroll
        for (int t = 0; t < 2; ++t) {
            int pair = tid + t * 256;
            int jr = pair >> 4, ic = pair & 15;
            float ph[HEADS];
#pragma unroll
            for (int hh = 0; hh < HEADS; ++hh) ph[hh] = lds_p[hh][jr][ic];
            float mix[HEADS];
            float mean = 0.0f;
#pragma unroll
            for (int g = 0; g < HEADS; ++g) {
                float sg = 0.0f;
#pragma unroll
                for (int hh = 0; hh < HEADS; ++hh) sg = fmaf(ph[hh], lds_W[hh][g], sg);
                mix[g] = sg; mean += sg;
            }
            mean *= 0.125f;
            float m2 = 0.0f;
#pragma unroll
            for (int g = 0; g < HEADS; ++g) { float d = mix[g] - mean; m2 = fmaf(d, d, m2); }
            float rstd = rsqrtf(m2 * 0.125f + 1e-5f);
#pragma unroll
            for (int g = 0; g < HEADS; ++g)
                lds_a[g][ic][jr] =
                    (_Float16)((mix[g] - mean) * rstd * lds_gam[g] + lds_bet[g]);
        }
        __syncthreads();

        // A fragment of normalized attn for this wave's head g (=h): rows i, K = j local
        {
            const _Float16* sa = &lds_a[h][lane & 15][(lane >> 4) << 4];
            V16U u;
            u.h[0] = *(const v8h*)(sa);
            u.h[1] = *(const v8h*)(sa + 8);

            // ensure the async V tile has landed in LDS
            asm volatile("s_wait_asynccnt 0x0" ::: "memory");

            int n15 = lane & 15;
            int kb = (lane >> 4) << 4;                 // 0 or 16
#pragma unroll
            for (int dt = 0; dt < 4; ++dt) {
                v16h bv;
#pragma unroll
                for (int t = 0; t < 16; ++t) bv[t] = lds_v[h][kb + t][dt * 16 + n15];
                accd[dt] = wmma_f16(u.v, bv, accd[dt]);
            }
        }
        __syncthreads();
    }

    // epilogue: out[b, i, h*64 + d] (f16 staging for the final projection GEMM)
    int ibase = i0 + ((lane >> 4) << 3);
#pragma unroll
    for (int dt = 0; dt < 4; ++dt)
#pragma unroll
        for (int r = 0; r < 8; ++r)
            attn_out[((size_t)b * SEQ + ibase + r) * DIMC + h * DHEAD + dt * 16 + (lane & 15)] =
                (_Float16)accd[dt][r];
}

// ---------------- output projection: attn_out[8192,512] @ w_out + b_out -> f32 ----
__global__ void out_gemm(const _Float16* __restrict__ ah, const _Float16* __restrict__ wT,
                         const float* __restrict__ bias, float* __restrict__ out) {
    int wave = threadIdx.x >> 5;
    int lane = threadIdx.x & 31;
    int m0 = blockIdx.x * 16;
    int n0 = wave * 64;                 // 8 waves x 64 = 512 columns
    v8f acc[4] = {{}, {}, {}, {}};
#pragma unroll 2
    for (int k0 = 0; k0 < DIMC; k0 += 32) {
        v16h a = load_frag_a(ah, m0, k0, DIMC);
#pragma unroll
        for (int dt = 0; dt < 4; ++dt) {
            v16h b = load_frag_bT(wT, k0, n0 + dt * 16, DIMC);
            acc[dt] = wmma_f16(a, b, acc[dt]);
        }
    }
    int mbase = m0 + ((lane >> 4) << 3);
#pragma unroll
    for (int dt = 0; dt < 4; ++dt) {
        int col = n0 + dt * 16 + (lane & 15);
        float bv = bias[col];
#pragma unroll
        for (int r = 0; r < 8; ++r)
            out[(size_t)(mbase + r) * DIMC + col] = acc[dt][r] + bv;
    }
}

extern "C" void kernel_launch(void* const* d_in, const int* in_sizes, int n_in,
                              void* d_out, int out_size, void* d_ws, size_t ws_size,
                              hipStream_t stream) {
    const float* x      = (const float*)d_in[0];
    const float* w_qkv  = (const float*)d_in[1];
    const float* reattn = (const float*)d_in[2];
    const float* ln_g   = (const float*)d_in[3];
    const float* ln_b   = (const float*)d_in[4];
    const float* w_out  = (const float*)d_in[5];
    const float* b_out  = (const float*)d_in[6];
    float* out = (float*)d_out;

    const size_t NHD = (size_t)BATCH * HEADS * SEQ * DHEAD;   // 4,194,304
    _Float16* ws    = (_Float16*)d_ws;
    _Float16* xh    = ws;                                     // [8192,512]
    _Float16* wqkvT = xh + (size_t)MTOT * DIMC;               // [1536,512]
    _Float16* woutT = wqkvT + (size_t)DIMC * QKVN;            // [512,512]
    _Float16* Qb    = woutT + (size_t)DIMC * DIMC;            // [B,H,N,D]
    _Float16* Kb    = Qb + NHD;
    _Float16* Vb    = Kb + NHD;
    _Float16* attnb = Vb + NHD;                               // [8192,512]
    size_t needed = (size_t)(attnb + (size_t)MTOT * DIMC - ws) * sizeof(_Float16);
    if (ws_size < needed) return;

    int nx = MTOT * DIMC;
    cvt_f16<<<(nx + 255) / 256, 256, 0, stream>>>(x, xh, nx);
    tcvt_f16<<<(DIMC * QKVN + 255) / 256, 256, 0, stream>>>(w_qkv, wqkvT, DIMC, QKVN);
    tcvt_f16<<<(DIMC * DIMC + 255) / 256, 256, 0, stream>>>(w_out, woutT, DIMC, DIMC);

    qkv_gemm<<<dim3(MTOT / 16, QKVN / 512), 256, 0, stream>>>(xh, wqkvT, Qb, Kb, Vb);
    reattn_core<<<dim3(SEQ / 16, BATCH), 256, 0, stream>>>(Qb, Kb, Vb, reattn, ln_g, ln_b, attnb);
    out_gemm<<<dim3(MTOT / 16, 1), 256, 0, stream>>>(attnb, woutT, b_out, out);
}
